// MaskedMHSA2d_71107478552971
// MI455X (gfx1250) — matrix operand order repaired and checked by reading, not compiled, not run on hardware
//
#include <hip/hip_runtime.h>

#define BATCH 8
#define CHANNELS 512
#define NUM_HEADS 8
#define HEAD_DIM 64
#define SEQ 1024
#define ATTN_SCALE 0.125f  // 1/sqrt(64)

typedef __attribute__((ext_vector_type(16))) __bf16 bf16x16;
typedef __attribute__((ext_vector_type(8)))  float  f32x8;
typedef __attribute__((ext_vector_type(4)))  int    i32x4;

static __device__ __forceinline__ f32x8 wmma_bf16(bf16x16 a, bf16x16 b, f32x8 c) {
  // D = A(16x32 bf16) * B(32x16 bf16) + C(16x16 f32)
  return __builtin_amdgcn_wmma_f32_16x16x32_bf16(false, a, false, b, (short)0, c,
                                                 false, false);
}

// A-fragment 16-bit layout (16x32): lane L holds row M=L&15; half i is
// K = i + hi*16, hi = L>=16.
// B-fragment 16-bit layout (32x16): lane L holds col N=L&15; half i is
// K = i + hi*16.

#if __has_builtin(__builtin_amdgcn_global_load_async_to_lds_b128) && \
    __has_builtin(__builtin_amdgcn_s_wait_asynccnt)
#define HAVE_ASYNC_LDS 1
#else
#define HAVE_ASYNC_LDS 0
#endif

static __device__ __forceinline__ void copy16_to_lds(const float* g, float* l) {
#if HAVE_ASYNC_LDS
  // signature (from hipcc diagnostic): (v4i AS1*, v4i AS3*, imm offset, imm cpol)
  __builtin_amdgcn_global_load_async_to_lds_b128(
      (__attribute__((address_space(1))) i32x4*)g,
      (__attribute__((address_space(3))) i32x4*)l, 0, 0);
#else
  *(float4*)l = *(const float4*)g;  // global_load_b128 + ds_store_b128
#endif
}

static __device__ __forceinline__ void wait_async_le4() {
#if HAVE_ASYNC_LDS
  __builtin_amdgcn_s_wait_asynccnt(4);
#endif
}
static __device__ __forceinline__ void wait_async_le0() {
#if HAVE_ASYNC_LDS
  __builtin_amdgcn_s_wait_asynccnt(0);
#endif
}

// ---------------------------------------------------------------------------
// QKV projection: qkv[b,o,s] = sum_c w[o,c]*x[b,c,s] + bias[o]
// Block (4 waves) computes a 64(o) x 64(s) tile; the x k-slab (32x64 f32) is
// staged into LDS with double-buffered async-to-LDS copies and shared by all
// waves. Output written bf16 as [b,h,s,d]; Q pre-scaled by 1/sqrt(head_dim).
// ---------------------------------------------------------------------------
__global__ __launch_bounds__(128)
void qkv_kernel(const float* __restrict__ x, const float* __restrict__ w,
                const float* __restrict__ bias,
                __bf16* __restrict__ qo, __bf16* __restrict__ ko,
                __bf16* __restrict__ vo) {
  __shared__ float xs[2][32 * 64];  // double-buffered k-slab, 8KB each

  const int lane = threadIdx.x & 31;
  const int wid  = threadIdx.x >> 5;
  const int bx    = blockIdx.x;
  const int sblk  = bx & 15;              // 16 blocks of 64 along S
  const int otile = (bx >> 4) % 24;       // 24 tiles of 64 along O=1536
  const int b     = bx / (16 * 24);
  const int o0 = otile * 64 + wid * 16;   // per-wave 16 output rows
  const int s0 = sblk * 64;
  const int n = lane & 15, hi = lane >> 4;

  // Per-thread async staging of x[b][k0+kk][s0..s0+63] (32 rows x 256B).
  // 128 threads x 4 chunks x 16B = 8KB. Fully coalesced row chunks.
  const int crow = threadIdx.x >> 4;        // 0..7
  const int ccol = (threadIdx.x & 15) * 4;  // float offset within row
  const float* xbase = x + (size_t)b * CHANNELS * SEQ + s0 + ccol;

  // prologue: issue k-slab 0 into buffer 0
#pragma unroll
  for (int rr = 0; rr < 4; ++rr) {
    const int kk = crow + rr * 8;
    copy16_to_lds(xbase + (size_t)kk * SEQ, &xs[0][kk * 64 + ccol]);
  }

  f32x8 acc[4] = {};
  const float* arow = w + (size_t)(o0 + n) * CHANNELS;

  for (int ks = 0; ks < 16; ++ks) {
    const int k0 = ks * 32;
    const float* lbuf = &xs[ks & 1][0];
    if (ks < 15) {  // issue next slab into the other buffer
      const int kn = k0 + 32;
#pragma unroll
      for (int rr = 0; rr < 4; ++rr) {
        const int kk = crow + rr * 8;
        copy16_to_lds(xbase + (size_t)(kn + kk) * SEQ,
                      &xs[(ks + 1) & 1][kk * 64 + ccol]);
      }
      wait_async_le4();  // current slab's 4 transfers complete (in-order)
    } else {
      wait_async_le0();
    }
    __syncthreads();

    bf16x16 a;
    const float* ap = arow + k0 + hi * 16;
#pragma unroll
    for (int i = 0; i < 8; ++i) { a[i] = (__bf16)ap[i]; a[i + 8] = (__bf16)ap[i + 8]; }
#pragma unroll
    for (int nt = 0; nt < 4; ++nt) {
      bf16x16 bb;
      const float* bp = lbuf + (size_t)(hi * 16) * 64 + nt * 16 + n;
#pragma unroll
      for (int i = 0; i < 16; ++i) bb[i] = (__bf16)bp[i * 64];
      acc[nt] = wmma_bf16(a, bb, acc[nt]);
    }
    __syncthreads();  // all waves done with lbuf before it is overwritten
  }

  const int part = o0 >> 9;               // 0=q 1=k 2=v (uniform in wave)
  const int h    = (o0 >> 6) & 7;         // head (uniform in wave)
  __bf16* dst = (part == 0) ? qo : (part == 1) ? ko : vo;
  const float mul = (part == 0) ? ATTN_SCALE : 1.0f;
#pragma unroll
  for (int nt = 0; nt < 4; ++nt) {
#pragma unroll
    for (int r = 0; r < 8; ++r) {
      const int o = o0 + r + hi * 8;
      const int s = s0 + nt * 16 + n;
      const int d = o & 63;
      const float val = (acc[nt][r] + bias[o]) * mul;
      dst[(((size_t)b * NUM_HEADS + h) * SEQ + s) * HEAD_DIM + d] = (__bf16)val;
    }
  }
}

// ---------------------------------------------------------------------------
// Flash attention: one wave handles 16 query rows for one (b,h).
// Online softmax with causal + key-padding masks; p tiles re-layouted
// D-format -> A-format via LDS; ctx accumulated 16x64 f32.
// Output ctx written bf16 as [b, s, c] (c = h*64 + d) for the proj GEMM.
// ---------------------------------------------------------------------------
__global__ __launch_bounds__(128)
void attn_kernel(const __bf16* __restrict__ q, const __bf16* __restrict__ kk,
                 const __bf16* __restrict__ vv,
                 const unsigned char* __restrict__ kpm,
                 __bf16* __restrict__ ctx) {
  __shared__ __bf16 lds_p[4][16 * 32];    // per-wave 16x32 probability tile
  const int lane = threadIdx.x & 31;
  const int wid  = threadIdx.x >> 5;
  const int wave = blockIdx.x * (blockDim.x >> 5) + wid;
  const int sblk = wave & 63;
  const int h    = (wave >> 6) & 7;
  const int b    = wave >> 9;
  const int s0 = sblk * 16;
  const int n = lane & 15, hi = lane >> 4;

  const size_t head_off = ((size_t)b * NUM_HEADS + h) * SEQ * HEAD_DIM;
  const __bf16* qb = q  + head_off;
  const __bf16* kb = kk + head_off;
  const __bf16* vb = vv + head_off;
  __bf16* pt = &lds_p[wid][0];

  const float NEG = -3.4028234663852886e38f;
  float mrun[8], lrun[8];
#pragma unroll
  for (int r = 0; r < 8; ++r) { mrun[r] = NEG; lrun[r] = 0.f; }
  f32x8 acc[4] = {};

  // Q A-fragments (rows s0..s0+15, full d=64 in two K=32 chunks) — loop invariant
  bf16x16 qa[2];
#pragma unroll
  for (int dk = 0; dk < 2; ++dk) {
    const __bf16* ap = qb + (size_t)(s0 + n) * HEAD_DIM + dk * 32 + hi * 16;
#pragma unroll
    for (int i = 0; i < 8; ++i) { qa[dk][i] = ap[i]; qa[dk][i + 8] = ap[i + 8]; }
  }

  const int tsteps = (s0 + 16 + 31) >> 5;   // causal: stop at diagonal block
  for (int tt = 0; tt < tsteps; ++tt) {
    const int t0 = tt * 32;
    // ---- scores: two 16x16 tiles (cols t0..t0+15 and t0+16..t0+31) ----
    f32x8 sc0 = {}, sc1 = {};
#pragma unroll
    for (int dk = 0; dk < 2; ++dk) {
      const int d0 = dk * 32;
      bf16x16 b0, b1;
      const __bf16* kp0 = kb + (size_t)(t0 + n) * HEAD_DIM + d0 + hi * 16;
      const __bf16* kp1 = kb + (size_t)(t0 + 16 + n) * HEAD_DIM + d0 + hi * 16;
#pragma unroll
      for (int i = 0; i < 16; ++i) { b0[i] = kp0[i]; b1[i] = kp1[i]; }
      sc0 = wmma_bf16(qa[dk], b0, sc0);
      sc1 = wmma_bf16(qa[dk], b1, sc1);
    }
    // ---- masks + online softmax (rows live in 16-lane halves) ----
    const int c0 = t0 + n, c1 = t0 + 16 + n;
    const bool km0 = kpm[(size_t)b * SEQ + c0] != 0;
    const bool km1 = kpm[(size_t)b * SEQ + c1] != 0;
    float p0[8], p1[8], alpha[8];
#pragma unroll
    for (int r = 0; r < 8; ++r) {
      const int row = s0 + r + hi * 8;
      float v0 = sc0[r], v1 = sc1[r];
      if (c0 > row || km0) v0 = NEG;
      if (c1 > row || km1) v1 = NEG;
      float mx = fmaxf(v0, v1);
      mx = fmaxf(mx, __shfl_xor(mx, 1, 32));
      mx = fmaxf(mx, __shfl_xor(mx, 2, 32));
      mx = fmaxf(mx, __shfl_xor(mx, 4, 32));
      mx = fmaxf(mx, __shfl_xor(mx, 8, 32));
      const float mnew = fmaxf(mrun[r], mx);
      alpha[r] = __expf(mrun[r] - mnew);
      mrun[r]  = mnew;
      p0[r] = __expf(v0 - mnew);
      p1[r] = __expf(v1 - mnew);
      float rs = p0[r] + p1[r];
      rs += __shfl_xor(rs, 1, 32);
      rs += __shfl_xor(rs, 2, 32);
      rs += __shfl_xor(rs, 4, 32);
      rs += __shfl_xor(rs, 8, 32);
      lrun[r] = lrun[r] * alpha[r] + rs;
    }
#pragma unroll
    for (int nt = 0; nt < 4; ++nt)
#pragma unroll
      for (int r = 0; r < 8; ++r) acc[nt][r] = acc[nt][r] * alpha[r];

    // ---- D-layout -> A-layout via LDS (same-wave LDS is in-order) ----
    asm volatile("" ::: "memory");
#pragma unroll
    for (int r = 0; r < 8; ++r) {
      pt[(r + hi * 8) * 32 + n]      = (__bf16)p0[r];
      pt[(r + hi * 8) * 32 + n + 16] = (__bf16)p1[r];
    }
    asm volatile("" ::: "memory");
    bf16x16 pa;
    {
      const __bf16* pp = pt + n * 32 + hi * 16;
#pragma unroll
      for (int i = 0; i < 8; ++i) { pa[i] = pp[i]; pa[i + 8] = pp[i + 8]; }
    }
    asm volatile("" ::: "memory");
    // ---- ctx += P * V ----
#pragma unroll
    for (int nt = 0; nt < 4; ++nt) {
      bf16x16 bb;
      const __bf16* vp = vb + (size_t)(t0 + hi * 16) * HEAD_DIM + nt * 16 + n;
#pragma unroll
      for (int i = 0; i < 16; ++i) bb[i] = vp[(size_t)i * HEAD_DIM];
      acc[nt] = wmma_bf16(pa, bb, acc[nt]);
    }
  }

  // ---- epilogue: normalize, write ctx[b][s][c] bf16 ----
#pragma unroll
  for (int nt = 0; nt < 4; ++nt) {
#pragma unroll
    for (int r = 0; r < 8; ++r) {
      const int s = s0 + r + hi * 8;
      const int c = h * HEAD_DIM + nt * 16 + n;
      const float inv = (lrun[r] > 0.f) ? 1.f / lrun[r] : 0.f;
      ctx[((size_t)b * SEQ + s) * CHANNELS + c] = (__bf16)(acc[nt][r] * inv);
    }
  }
}

// ---------------------------------------------------------------------------
// Output projection: out[b,o,s] = sum_c proj_w[o,c]*ctx[b,s,c] + proj_b[o]
// B fragments read contiguous 32B runs of bf16 ctx; A converted f32->bf16.
// ---------------------------------------------------------------------------
__global__ __launch_bounds__(128)
void proj_kernel(const __bf16* __restrict__ cx, const float* __restrict__ w,
                 const float* __restrict__ bias, float* __restrict__ out) {
  const int lane = threadIdx.x & 31;
  const int wave = blockIdx.x * (blockDim.x >> 5) + (threadIdx.x >> 5);
  const int sblk  = wave & 15;
  const int otile = (wave >> 4) & 31;
  const int b     = wave >> 9;
  const int o0 = otile * 16, s0 = sblk * 64;
  const int n = lane & 15, hi = lane >> 4;

  f32x8 acc[4] = {};
  const float* arow = w + (size_t)(o0 + n) * CHANNELS;
  for (int k0 = 0; k0 < CHANNELS; k0 += 32) {
    bf16x16 a;
    const float* ap = arow + k0 + hi * 16;
#pragma unroll
    for (int i = 0; i < 8; ++i) { a[i] = (__bf16)ap[i]; a[i + 8] = (__bf16)ap[i + 8]; }
#pragma unroll
    for (int nt = 0; nt < 4; ++nt) {
      const int s = s0 + nt * 16 + n;
      const __bf16* bp = cx + ((size_t)b * SEQ + s) * CHANNELS + k0 + hi * 16;
      bf16x16 bb;
#pragma unroll
      for (int i = 0; i < 16; ++i) bb[i] = bp[i];
      acc[nt] = wmma_bf16(a, bb, acc[nt]);
    }
  }
#pragma unroll
  for (int nt = 0; nt < 4; ++nt) {
#pragma unroll
    for (int r = 0; r < 8; ++r) {
      const int o = o0 + r + hi * 8;
      const int s = s0 + nt * 16 + n;
      out[((size_t)b * CHANNELS + o) * SEQ + s] = acc[nt][r] + bias[o];
    }
  }
}

// ---------------------------------------------------------------------------
extern "C" void kernel_launch(void* const* d_in, const int* in_sizes, int n_in,
                              void* d_out, int out_size, void* d_ws, size_t ws_size,
                              hipStream_t stream) {
  const float* x          = (const float*)d_in[0];
  const unsigned char* am = (const unsigned char*)d_in[1];  // bool mask
  const float* qkv_w      = (const float*)d_in[2];
  const float* qkv_b      = (const float*)d_in[3];
  const float* proj_w     = (const float*)d_in[4];
  const float* proj_b     = (const float*)d_in[5];
  float* out = (float*)d_out;

  const size_t HS = (size_t)BATCH * NUM_HEADS * SEQ * HEAD_DIM;  // 4 Mi elems
  __bf16* qws = (__bf16*)d_ws;
  __bf16* kws = qws + HS;
  __bf16* vws = kws + HS;
  __bf16* cws = vws + HS;   // ctx [b,s,c], 4 Mi elems; total ws = 32 MB

  {
    const int blocks = BATCH * 24 * 16;                // 3072 blocks, 4 waves ea.
    qkv_kernel<<<dim3(blocks), dim3(128), 0, stream>>>(x, qkv_w, qkv_b,
                                                       qws, kws, vws);
  }
  {
    const int waves = BATCH * NUM_HEADS * (SEQ / 16);  // 4096
    attn_kernel<<<dim3(waves / 4), dim3(128), 0, stream>>>(qws, kws, vws, am, cws);
  }
  {
    const int waves = BATCH * 32 * 16;                 // 4096
    proj_kernel<<<dim3(waves / 4), dim3(128), 0, stream>>>(cws, proj_w, proj_b, out);
  }
}
